// PureGAT_39676907888758
// MI455X (gfx1250) — compile-verified, matrix operand order checked by reading, not run on hardware
//
#include <hip/hip_runtime.h>
#include <hip/hip_bf16.h>
#include <math.h>

typedef __attribute__((ext_vector_type(16))) __bf16 v16bf;
typedef __attribute__((ext_vector_type(8)))  float  v8f;

#define NN   50000
#define NE   800000
#define NB   64
#define HIDD 128

// ---------- monotonic float <-> uint punning for atomicMax-based segment max ----------
__device__ __forceinline__ unsigned fpun(float x) {
  unsigned u = __float_as_uint(x);
  return (u & 0x80000000u) ? ~u : (u | 0x80000000u);
}
__device__ __forceinline__ float funpun(unsigned u) {
  u = (u & 0x80000000u) ? (u & 0x7FFFFFFFu) : ~u;
  return __uint_as_float(u);
}

// ---------------------------------------------------------------------------
// WMMA GEMM:  O[M,Nc] = act(A[M,K] @ B[K,Nc] (+ bias)), M % 16 == 0.
// K compile-time (64/128) -> fully unrolled k loop.
// One wave computes a 64x16 output strip (4 M-tiles x 1 N-tile): the B
// fragment is loaded once per k-step and reused by 4 WMMAs.  The wave id is
// built from readfirstlane so all tile guards are *scalar* branches: EXEC is
// never modified (WMMA requires EXEC all-ones) and no saveexec churn.
// VGPR layouts per CDNA5 ISA 7.12.2:
//   A 16x32 bf16: lane = M%16, half s = lane/16; vgpr j pair holds
//                 K = (j>=4)*16 + s*8 + (j&3)*2  (+0,+1)
//   B 32x16 bf16: symmetric, lane = N%16
//   C/D 16x16 f32: lane = n + 16*(m/8), vgpr v = m%8
// ---------------------------------------------------------------------------
template <int K>
__global__ __launch_bounds__(256)
void k_gemm_wmma(const float* __restrict__ A, const float* __restrict__ B,
                 const float* __restrict__ bias, float* __restrict__ O,
                 int M, int Nc, int act) {
  // wave-uniform (scalar) wave id: blockDim.x == 256 -> 8 waves per block
  const int wib  = __builtin_amdgcn_readfirstlane(threadIdx.x >> 5);
  const int wave = blockIdx.x * 8 + wib;
  const int lane = threadIdx.x & 31;

  const int tilesN = Nc >> 4;
  const int tilesM = M >> 4;
  const int gm     = wave / tilesN;          // group of 4 M-tiles (scalar)
  const int tn     = wave - gm * tilesN;     // N tile (scalar)
  if (gm * 4 >= tilesM) return;
  const int tm0 = gm * 4;

  const int s   = lane >> 4;                 // half-wave select
  const int r15 = lane & 15;
  const int col = tn * 16 + r15;

  v8f acc0 = {}, acc1 = {}, acc2 = {}, acc3 = {};

#pragma unroll
  for (int k0 = 0; k0 < K; k0 += 32) {
    v16bf bfrag;
#pragma unroll
    for (int j = 0; j < 8; ++j) {
      const int kk = k0 + ((j >> 2) << 4) + (s << 3) + ((j & 3) << 1);
      bfrag[2 * j]     = (__bf16)B[kk * Nc + col];
      bfrag[2 * j + 1] = (__bf16)B[(kk + 1) * Nc + col];
    }
#pragma unroll
    for (int t = 0; t < 4; ++t) {
      if (tm0 + t < tilesM) {                // scalar compare -> s_cbranch
        const int row = (tm0 + t) * 16 + r15;
        v16bf afrag;
#pragma unroll
        for (int j = 0; j < 8; ++j) {
          const int kk = k0 + ((j >> 2) << 4) + (s << 3) + ((j & 3) << 1);
          const float2 a2 = *(const float2*)(A + (size_t)row * K + kk);
          afrag[2 * j]     = (__bf16)a2.x;
          afrag[2 * j + 1] = (__bf16)a2.y;
        }
        v8f& acc = (t == 0) ? acc0 : (t == 1) ? acc1 : (t == 2) ? acc2 : acc3;
        acc = __builtin_amdgcn_wmma_f32_16x16x32_bf16(
            false, afrag, false, bfrag, (short)0, acc, false, false);
      }
    }
  }

  const float bi = bias ? bias[col] : 0.f;
#pragma unroll
  for (int t = 0; t < 4; ++t) {
    if (tm0 + t < tilesM) {
      const v8f& acc = (t == 0) ? acc0 : (t == 1) ? acc1 : (t == 2) ? acc2 : acc3;
#pragma unroll
      for (int v = 0; v < 8; ++v) {
        const int m = (tm0 + t) * 16 + s * 8 + v;
        float r = acc[v] + bi;
        if (act) r = fmaxf(r, 0.f);
        O[(size_t)m * Nc + col] = r;
      }
    }
  }
}

// ---------------------------------------------------------------------------
// Fold attention-edge weights:  wae[d,h] = sum_c We[d, h*C+c] * ae[h,c]
// (collapses the [E,16]@[16,128] GEMM that is only reduced against ae)
// ---------------------------------------------------------------------------
__global__ void k_wae(const float* __restrict__ We, const float* __restrict__ ae,
                      float* __restrict__ wae, int H) {
  const int t = threadIdx.x;
  if (t >= 16 * H) return;
  const int C = HIDD / H;
  const int d = t / H, h = t - (t / H) * H;
  float sacc = 0.f;
  for (int c = 0; c < C; ++c) sacc += We[d * HIDD + h * C + c] * ae[h * C + c];
  wae[d * H + h] = sacc;
}

// ---------------------------------------------------------------------------
// Per-node attention logits: al_src/al_dst[n,h] = sum_c xp[n,h,c]*a[h,c]
// One wave per node, 4 floats/lane, butterfly reduce within head groups.
// ---------------------------------------------------------------------------
template <int H>
__global__ __launch_bounds__(256)
void k_node_attn(const float* __restrict__ xp, const float* __restrict__ a_s,
                 const float* __restrict__ a_d, float* __restrict__ alsrc,
                 float* __restrict__ aldst) {
  const int wid  = (blockIdx.x * blockDim.x + threadIdx.x) >> 5;
  const int lane = threadIdx.x & 31;
  if (wid >= NN) return;
  const float4 x4 = *(const float4*)(xp + (size_t)wid * HIDD + lane * 4);
  const float4 s4 = *(const float4*)(a_s + lane * 4);
  const float4 d4 = *(const float4*)(a_d + lane * 4);
  float ps = x4.x * s4.x + x4.y * s4.y + x4.z * s4.z + x4.w * s4.w;
  float pd = x4.x * d4.x + x4.y * d4.y + x4.z * d4.z + x4.w * d4.w;
  constexpr int gl = 32 / H;  // lanes per head
#pragma unroll
  for (int m = 1; m < gl; m <<= 1) {
    ps += __shfl_xor(ps, m, 32);
    pd += __shfl_xor(pd, m, 32);
  }
  if ((lane & (gl - 1)) == 0) {
    const int h = lane / gl;
    alsrc[wid * H + h] = ps;
    aldst[wid * H + h] = pd;
  }
}

// ---------------------------------------------------------------------------
// Edge pass 1: alpha = leaky_relu(al_src[src]+al_dst[dst]+ea.wae, 0.2);
// store alpha, atomic segment-max into umax[dst,h] (uint-punned).
// ---------------------------------------------------------------------------
template <int H>
__global__ __launch_bounds__(256)
void k_edge1(const int* __restrict__ src, const int* __restrict__ dst,
             const float* __restrict__ ea, const float* __restrict__ alsrc,
             const float* __restrict__ aldst, const float* __restrict__ wae,
             float* __restrict__ alpha, unsigned* __restrict__ umax) {
  const int e = blockIdx.x * blockDim.x + threadIdx.x;
  if (e >= NE) return;
  const int sI = src[e], dI = dst[e];
  float eav[16];
  const float4* ep = (const float4*)(ea + (size_t)e * 16);
#pragma unroll
  for (int q = 0; q < 4; ++q) {
    const float4 v = ep[q];
    eav[4 * q + 0] = v.x; eav[4 * q + 1] = v.y;
    eav[4 * q + 2] = v.z; eav[4 * q + 3] = v.w;
  }
#pragma unroll
  for (int h = 0; h < H; ++h) {
    float ale = 0.f;
#pragma unroll
    for (int i = 0; i < 16; ++i) ale += eav[i] * wae[i * H + h];
    float a = alsrc[sI * H + h] + aldst[dI * H + h] + ale;
    a = (a > 0.f) ? a : 0.2f * a;
    alpha[(size_t)e * H + h] = a;
    atomicMax(umax + dI * H + h, fpun(a));
  }
}

// ---------------------------------------------------------------------------
// Edge pass 2: ex = exp(alpha - max[dst]); alpha <- ex; segment-sum into nsum.
// ---------------------------------------------------------------------------
template <int H>
__global__ __launch_bounds__(256)
void k_edge2(const int* __restrict__ dst, float* __restrict__ alpha,
             const unsigned* __restrict__ umax, float* __restrict__ nsum) {
  const int i = blockIdx.x * blockDim.x + threadIdx.x;
  if (i >= NE * H) return;
  const int e = i / H, h = i - e * H;
  const int d = dst[e];
  const float m  = funpun(umax[d * H + h]);
  const float ex = __expf(alpha[i] - m);
  alpha[i] = ex;
  atomicAdd(nsum + d * H + h, ex);
}

// ---------------------------------------------------------------------------
// Edge pass 3: out[dst] += xp[src] * (ex / (nsum[dst]+1e-16)).
// One wave per edge, float4 per lane, f32 atomics (L2-resident on MI455X:
// all node arrays fit in the 192MB L2).
// ---------------------------------------------------------------------------
template <int H>
__global__ __launch_bounds__(256)
void k_scatter(const int* __restrict__ src, const int* __restrict__ dst,
               const float* __restrict__ xp, const float* __restrict__ alpha,
               const float* __restrict__ nsum, float* __restrict__ out) {
  const int e    = (blockIdx.x * blockDim.x + threadIdx.x) >> 5;
  const int lane = threadIdx.x & 31;
  if (e >= NE) return;
  const int sI = src[e], dI = dst[e];
  if (e + 256 < NE)  // warm L2 ahead of time (global_prefetch_b8)
    __builtin_prefetch(xp + (size_t)src[e + 256] * HIDD + lane * 4, 0, 0);
  constexpr int C = HIDD / H;
  const int h = (lane * 4) / C;
  const float w = alpha[(size_t)e * H + h] / (nsum[dI * H + h] + 1e-16f);
  const float4 v = *(const float4*)(xp + (size_t)sI * HIDD + lane * 4);
  float* o = out + (size_t)dI * HIDD + lane * 4;
  atomicAdd(o + 0, v.x * w);
  atomicAdd(o + 1, v.y * w);
  atomicAdd(o + 2, v.z * w);
  atomicAdd(o + 3, v.w * w);
}

// ---------------------------------------------------------------------------
// Per-node epilogue: h <- elu(layernorm(out + b; g, be)) + h   (residual)
// ---------------------------------------------------------------------------
__global__ __launch_bounds__(256)
void k_finish(const float* __restrict__ out, const float* __restrict__ b,
              const float* __restrict__ g, const float* __restrict__ be,
              float* __restrict__ h) {
  const int wid  = (blockIdx.x * blockDim.x + threadIdx.x) >> 5;
  const int lane = threadIdx.x & 31;
  if (wid >= NN) return;
  float4 v = *(const float4*)(out + (size_t)wid * HIDD + lane * 4);
  const float4 bb = *(const float4*)(b + lane * 4);
  v.x += bb.x; v.y += bb.y; v.z += bb.z; v.w += bb.w;
  float sacc = v.x + v.y + v.z + v.w;
#pragma unroll
  for (int m = 1; m < 32; m <<= 1) sacc += __shfl_xor(sacc, m, 32);
  const float mean = sacc * (1.f / HIDD);
  const float dx = v.x - mean, dy = v.y - mean, dz = v.z - mean, dw = v.w - mean;
  float vacc = dx * dx + dy * dy + dz * dz + dw * dw;
#pragma unroll
  for (int m = 1; m < 32; m <<= 1) vacc += __shfl_xor(vacc, m, 32);
  const float inv = rsqrtf(vacc * (1.f / HIDD) + 1e-5f);
  const float4 gg = *(const float4*)(g + lane * 4);
  const float4 ee = *(const float4*)(be + lane * 4);
  float y0 = dx * inv * gg.x + ee.x;
  float y1 = dy * inv * gg.y + ee.y;
  float y2 = dz * inv * gg.z + ee.z;
  float y3 = dw * inv * gg.w + ee.w;
  y0 = (y0 > 0.f) ? y0 : __expf(y0) - 1.f;
  y1 = (y1 > 0.f) ? y1 : __expf(y1) - 1.f;
  y2 = (y2 > 0.f) ? y2 : __expf(y2) - 1.f;
  y3 = (y3 > 0.f) ? y3 : __expf(y3) - 1.f;
  float4* hp = (float4*)(h + (size_t)wid * HIDD + lane * 4);
  float4 r = *hp;
  r.x += y0; r.y += y1; r.z += y2; r.w += y3;
  *hp = r;
}

// ---------------------------------------------------------------------------
// Gate scalar: gate[n] = t[n,0:64].gW2 + gb2 ; atomic per-graph max.
// ---------------------------------------------------------------------------
__global__ __launch_bounds__(256)
void k_gate(const float* __restrict__ t, const float* __restrict__ gW2,
            const float* __restrict__ gb2, const int* __restrict__ batch,
            float* __restrict__ gate, unsigned* __restrict__ gmax) {
  const int wid  = (blockIdx.x * blockDim.x + threadIdx.x) >> 5;
  const int lane = threadIdx.x & 31;
  if (wid >= NN) return;
  float p = t[(size_t)wid * 64 + lane] * gW2[lane] +
            t[(size_t)wid * 64 + 32 + lane] * gW2[32 + lane];
#pragma unroll
  for (int m = 1; m < 32; m <<= 1) p += __shfl_xor(p, m, 32);
  if (lane == 0) {
    const float gv = p + gb2[0];
    gate[wid] = gv;
    atomicMax(gmax + batch[wid], fpun(gv));
  }
}

__global__ __launch_bounds__(256)
void k_gexp(const float* __restrict__ gate, const unsigned* __restrict__ gmax,
            const int* __restrict__ batch, float* __restrict__ ge,
            float* __restrict__ gden) {
  const int i = blockIdx.x * blockDim.x + threadIdx.x;
  if (i >= NN) return;
  const int b = batch[i];
  const float ex = __expf(gate[i] - funpun(gmax[b]));
  ge[i] = ex;
  atomicAdd(gden + b, ex);
}

__global__ __launch_bounds__(256)
void k_pool(const float* __restrict__ h, const float* __restrict__ ge,
            const float* __restrict__ gden, const int* __restrict__ batch,
            float* __restrict__ pooled) {
  const int wid  = (blockIdx.x * blockDim.x + threadIdx.x) >> 5;
  const int lane = threadIdx.x & 31;
  if (wid >= NN) return;
  const int b = batch[wid];
  const float att = ge[wid] / (gden[b] + 1e-16f);
  const float4 v = *(const float4*)(h + (size_t)wid * HIDD + lane * 4);
  float* p = pooled + (size_t)b * HIDD + lane * 4;
  atomicAdd(p + 0, v.x * att);
  atomicAdd(p + 1, v.y * att);
  atomicAdd(p + 2, v.z * att);
  atomicAdd(p + 3, v.w * att);
}

// Final tiny layer: out[64,6] = ctmp[64,64] @ cW2[64,6] + cb2
__global__ __launch_bounds__(128)
void k_cls2(const float* __restrict__ ctmp, const float* __restrict__ cW2,
            const float* __restrict__ cb2, float* __restrict__ out) {
  const int i = blockIdx.x * blockDim.x + threadIdx.x;
  if (i >= NB * 6) return;
  const int b = i / 6, c = i - b * 6;
  float sacc = cb2[c];
  for (int k = 0; k < 64; ++k) sacc += ctmp[b * 64 + k] * cW2[k * 6 + c];
  out[i] = sacc;
}

// ---------------------------------------------------------------------------
static inline void launch_gemm(const float* A, const float* B, const float* bias,
                               float* O, int M, int K, int Nc, int act,
                               hipStream_t s) {
  const long long tilesM = M / 16;
  const long long groups = (tilesM + 3) / 4;
  const long long waves  = groups * (Nc / 16);
  const int blocks = (int)((waves * 32 + 255) / 256);
  if (K == 64)
    k_gemm_wmma<64><<<blocks, 256, 0, s>>>(A, B, bias, O, M, Nc, act);
  else
    k_gemm_wmma<128><<<blocks, 256, 0, s>>>(A, B, bias, O, M, Nc, act);
}

extern "C" void kernel_launch(void* const* d_in, const int* in_sizes, int n_in,
                              void* d_out, int out_size, void* d_ws, size_t ws_size,
                              hipStream_t stream) {
  const float* x     = (const float*)d_in[0];
  const int*   ei    = (const int*)d_in[1];
  const float* ea    = (const float*)d_in[2];
  const int*   batch = (const int*)d_in[3];
  const float* pW    = (const float*)d_in[4];
  const float* pb    = (const float*)d_in[5];
  const int* src = ei;
  const int* dst = ei + NE;
  auto P = [&](int i) { return (const float*)d_in[i]; };
  const float *gW1 = P(30), *gb1 = P(31), *gW2 = P(32), *gb2 = P(33);
  const float *cW1 = P(34), *cb1 = P(35), *cW2 = P(36), *cb2 = P(37);

  // ---- workspace carve-out ----
  char* ws = (char*)d_ws;
  size_t off = 0;
  auto carve = [&](size_t bytes) {
    void* p = ws + off;
    off += (bytes + 255) & ~(size_t)255;
    return p;
  };
  float*    h     = (float*)carve((size_t)NN * HIDD * 4);
  float*    xp    = (float*)carve((size_t)NN * HIDD * 4);  // reused as gate-hidden t
  float*    outb  = (float*)carve((size_t)NN * HIDD * 4);
  float*    alsrc = (float*)carve((size_t)NN * 4 * 4);
  float*    aldst = (float*)carve((size_t)NN * 4 * 4);
  unsigned* umax  = (unsigned*)carve((size_t)NN * 4 * 4);
  float*    nsum  = (float*)carve((size_t)NN * 4 * 4);
  float*    alpha = (float*)carve((size_t)NE * 4 * 4);
  float*    wae   = (float*)carve(16 * 4 * 4);
  float*    gate  = (float*)carve((size_t)NN * 4);
  float*    ge    = (float*)carve((size_t)NN * 4);
  unsigned* gmax  = (unsigned*)carve(NB * 4);
  float*    gden  = (float*)carve(NB * 4);
  float*    pooled= (float*)carve((size_t)NB * HIDD * 4);
  float*    ctmp  = (float*)carve((size_t)NB * 64 * 4);

  const int nodeWaveBlocks = (NN * 32 + 255) / 256;
  const int edgeBlocks     = (NE + 255) / 256;
  const int edgeWaveBlocks = (NE * 32 + 255) / 256;

  // ---- 1. projection: h = relu(x @ pW + pb) ----
  launch_gemm(x, pW, pb, h, NN, 64, HIDD, 1, stream);

  // ---- 2. three GAT layers ----
  for (int l = 0; l < 3; ++l) {
    const int base = 6 + 8 * l;
    const float *W  = P(base + 0), *as_ = P(base + 1), *ad_ = P(base + 2),
                *ae_ = P(base + 3), *We = P(base + 4), *bl = P(base + 5),
                *gl = P(base + 6), *bel = P(base + 7);

    hipMemsetAsync(umax, 0, (size_t)NN * 4 * 4, stream);   // punned(-NaN) = 0
    hipMemsetAsync(nsum, 0, (size_t)NN * 4 * 4, stream);
    hipMemsetAsync(outb, 0, (size_t)NN * HIDD * 4, stream);

    launch_gemm(h, W, nullptr, xp, NN, HIDD, HIDD, 0, stream);   // xp = h@W
    if (l < 2) {
      k_wae<<<1, 64, 0, stream>>>(We, ae_, wae, 4);
      k_node_attn<4><<<nodeWaveBlocks, 256, 0, stream>>>(xp, as_, ad_, alsrc, aldst);
      k_edge1<4><<<edgeBlocks, 256, 0, stream>>>(src, dst, ea, alsrc, aldst, wae,
                                                 alpha, umax);
      k_edge2<4><<<(NE * 4 + 255) / 256, 256, 0, stream>>>(dst, alpha, umax, nsum);
      k_scatter<4><<<edgeWaveBlocks, 256, 0, stream>>>(src, dst, xp, alpha, nsum,
                                                       outb);
    } else {
      k_wae<<<1, 64, 0, stream>>>(We, ae_, wae, 1);
      k_node_attn<1><<<nodeWaveBlocks, 256, 0, stream>>>(xp, as_, ad_, alsrc, aldst);
      k_edge1<1><<<edgeBlocks, 256, 0, stream>>>(src, dst, ea, alsrc, aldst, wae,
                                                 alpha, umax);
      k_edge2<1><<<(NE + 255) / 256, 256, 0, stream>>>(dst, alpha, umax, nsum);
      k_scatter<1><<<edgeWaveBlocks, 256, 0, stream>>>(src, dst, xp, alpha, nsum,
                                                       outb);
    }
    k_finish<<<nodeWaveBlocks, 256, 0, stream>>>(outb, bl, gl, bel, h);
  }

  // ---- 3. global attention pooling ----
  hipMemsetAsync(gmax, 0, NB * 4, stream);
  hipMemsetAsync(gden, 0, NB * 4, stream);
  hipMemsetAsync(pooled, 0, (size_t)NB * HIDD * 4, stream);
  launch_gemm(h, gW1, gb1, xp, NN, HIDD, 64, 1, stream);  // t = relu(h@gW1+gb1)
  k_gate<<<nodeWaveBlocks, 256, 0, stream>>>(xp, gW2, gb2, batch, gate, gmax);
  k_gexp<<<(NN + 255) / 256, 256, 0, stream>>>(gate, gmax, batch, ge, gden);
  k_pool<<<nodeWaveBlocks, 256, 0, stream>>>(h, ge, gden, batch, pooled);

  // ---- 4. classifier ----
  launch_gemm(pooled, cW1, cb1, ctmp, NB, HIDD, 64, 1, stream);
  k_cls2<<<(NB * 6 + 127) / 128, 128, 0, stream>>>(ctmp, cW2, cb2, (float*)d_out);
}
